// CausalAttention_7619271983748
// MI455X (gfx1250) — matrix operand order repaired and checked by reading, not compiled, not run on hardware
//
#include <hip/hip_runtime.h>

typedef _Float16 h16;
typedef __attribute__((ext_vector_type(16))) _Float16 v16h;
typedef __attribute__((ext_vector_type(8)))  _Float16 v8h;
typedef __attribute__((ext_vector_type(4)))  _Float16 v4h;
typedef __attribute__((ext_vector_type(8)))  float    v8f;
typedef __attribute__((ext_vector_type(4)))  float    v4f;

#define WMMA_F32_F16(a, b, c) \
  __builtin_amdgcn_wmma_f32_16x16x32_f16(false, (a), false, (b), (short)0, (c), false, false)

// Async DMA: global -> LDS, 16B per lane, tracked by ASYNCcnt (CDNA5 §15.18 op 98).
__device__ __forceinline__ void async_copy_b128(const void* gaddr, void* lds_ptr) {
  unsigned lds_off = (unsigned)(unsigned long long)lds_ptr;  // low 32b = LDS offset
  asm volatile("global_load_async_to_lds_b128 %0, %1, off"
               :
               : "v"(lds_off), "v"((unsigned long long)gaddr)
               : "memory");
}
__device__ __forceinline__ void async_wait0() {
  asm volatile("s_wait_asynccnt 0" ::: "memory");
}
__device__ __forceinline__ void async_wait4() {  // allow one 4-instr quarter in flight
  asm volatile("s_wait_asynccnt 4" ::: "memory");
}

// A fragment (16x32 f16): lane 0-15 -> row M=lane, K halves {0..7,16..23};
// lane 16-31 -> row M=lane-16, K halves {8..15,24..31}. (ISA 7.12.2)
__device__ __forceinline__ v16h frag_a_load(const h16* base, int ld) {
  const int lane = threadIdx.x & 31;
  const h16* p = base + (lane & 15) * ld + ((lane >> 4) << 3);
  v8h lo = *(const v8h*)p;
  v8h hi = *(const v8h*)(p + 16);
  v16h r;
#pragma unroll
  for (int i = 0; i < 8; ++i) { r[i] = lo[i]; r[8 + i] = hi[i]; }
  return r;
}

// B fragment (32x16 f16): lane -> column N=lane%16; K halves {0..7,16..23} (lanes 0-15), {8..15,24..31} (lanes 16-31).
__device__ __forceinline__ v16h frag_b_load(const h16* base, int ld) {
  const int lane = threadIdx.x & 31;
  const h16* p = base + (lane & 15) + ((lane >> 4) << 3) * ld;
  v16h r;
#pragma unroll
  for (int i = 0; i < 8; ++i) { r[i] = p[i * ld]; r[8 + i] = p[(16 + i) * ld]; }
  return r;
}

// ---------------------------------------------------------------------------
// Kernel 1: fused fp32->f16 convert + QKV projection GEMM.
// Out[z] = X(8192x1024) * W[z](1024x1024), f16 result into workspace.
// Block: 256 threads = 8 waves; tile 128(M) x 128(N); K staged in chunks of 32.
// ---------------------------------------------------------------------------
__global__ __launch_bounds__(256) void qkv_proj(
    const float* __restrict__ X,
    const float* __restrict__ Wq, const float* __restrict__ Wk,
    const float* __restrict__ Wv,
    h16* __restrict__ Qh, h16* __restrict__ Kh, h16* __restrict__ Vh) {
  constexpr int LDA = 40;   // 32 + pad, halves
  constexpr int LDB = 136;  // 128 + pad, halves
  __shared__ __align__(16) h16 Ah[128 * LDA];
  __shared__ __align__(16) h16 Bh[32 * LDB];

  const float* W = (blockIdx.z == 0) ? Wq : (blockIdx.z == 1 ? Wk : Wv);
  h16* Out = (blockIdx.z == 0) ? Qh : (blockIdx.z == 1 ? Kh : Vh);

  const int m0 = blockIdx.x * 128, n0 = blockIdx.y * 128;
  const int t = threadIdx.x, wave = t >> 5, lane = t & 31;
  const int wm = wave & 3, wn = wave >> 2;  // 4x2 wave grid, 32x64 per wave

  v8f acc[2][4];
#pragma unroll
  for (int a = 0; a < 2; ++a)
#pragma unroll
    for (int b = 0; b < 4; ++b)
#pragma unroll
      for (int i = 0; i < 8; ++i) acc[a][b][i] = 0.0f;

  for (int k0 = 0; k0 < 1024; k0 += 32) {
    // Prefetch next K-chunk into GL2 while this one is consumed.
    if (k0 + 32 < 1024) {
      __builtin_prefetch(X + (size_t)(m0 + (t >> 3)) * 1024 + k0 + 32 + ((t & 7) << 2), 0, 3);
      __builtin_prefetch(W + (size_t)(k0 + 32 + (t >> 5)) * 1024 + n0 + ((t & 31) << 2), 0, 3);
    }
    __syncthreads();
    // Stage A: 128x32 fp32 -> f16 (1024 float4 / 256 threads)
#pragma unroll
    for (int i = 0; i < 4; ++i) {
      int idx = t + 256 * i;
      int r = idx >> 3, c = (idx & 7) << 2;
      v4f v = *(const v4f*)(X + (size_t)(m0 + r) * 1024 + k0 + c);
      v4h hv;
      hv[0] = (h16)v[0]; hv[1] = (h16)v[1]; hv[2] = (h16)v[2]; hv[3] = (h16)v[3];
      *(v4h*)(Ah + r * LDA + c) = hv;
    }
    // Stage B: 32x128 fp32 -> f16
#pragma unroll
    for (int i = 0; i < 4; ++i) {
      int idx = t + 256 * i;
      int r = idx >> 5, c = (idx & 31) << 2;
      v4f v = *(const v4f*)(W + (size_t)(k0 + r) * 1024 + n0 + c);
      v4h hv;
      hv[0] = (h16)v[0]; hv[1] = (h16)v[1]; hv[2] = (h16)v[2]; hv[3] = (h16)v[3];
      *(v4h*)(Bh + r * LDB + c) = hv;
    }
    __syncthreads();

    v16h a0 = frag_a_load(Ah + (wm * 32) * LDA, LDA);
    v16h a1 = frag_a_load(Ah + (wm * 32 + 16) * LDA, LDA);
#pragma unroll
    for (int fn = 0; fn < 4; ++fn) {
      v16h bb = frag_b_load(Bh + wn * 64 + fn * 16, LDB);
      acc[0][fn] = WMMA_F32_F16(a0, bb, acc[0][fn]);
      acc[1][fn] = WMMA_F32_F16(a1, bb, acc[1][fn]);
    }
  }

  const int hi8 = (lane >> 4) << 3, nl = lane & 15;
#pragma unroll
  for (int fm = 0; fm < 2; ++fm)
#pragma unroll
    for (int fn = 0; fn < 4; ++fn)
#pragma unroll
      for (int v = 0; v < 8; ++v) {
        int row = m0 + wm * 32 + fm * 16 + v + hi8;
        int col = n0 + wn * 64 + fn * 16 + nl;
        Out[(size_t)row * 1024 + col] = (h16)acc[fm][fn][v];
      }
}

// ---------------------------------------------------------------------------
// Kernel 2: causal flash attention over f16 Q/K/V, fp32 accumulate.
// Workgroup: 16 query rows, 8 waves; wave owns dims {32c : c%8==wave} (128 dims).
// KV block = 32 keys. K staged transposed in 256-dim quarters (reg-blocked
// transpose). V fetched by double-buffered async DMA quarters issued at block
// start, drained with partial s_wait_asynccnt while QK/softmax execute.
// ---------------------------------------------------------------------------
__global__ __launch_bounds__(256) void flash_attn(
    const h16* __restrict__ Qh, const h16* __restrict__ Kh,
    const h16* __restrict__ Vh, float* __restrict__ O) {
  __shared__ __align__(16) h16  Kbuf[256 * 32];    // 16KB, K^T quarter [d][kp]
  __shared__ __align__(16) h16  Vbuf[2][32 * 256]; // 2x16KB, V quarters [kp][d]
  __shared__ __align__(16) float Sc[16 * 32];      // raw QK^T scores
  __shared__ __align__(16) h16  Ph[16 * 32];       // probabilities (f16)
  __shared__ float mrow[16], lrow[16], crow[16];

  const int b = blockIdx.y;
  const int q0 = blockIdx.x * 16;
  const h16* Qb = Qh + (size_t)b * 2048 * 1024;
  const h16* Kb = Kh + (size_t)b * 2048 * 1024;
  const h16* Vb = Vh + (size_t)b * 2048 * 1024;
  float* Ob = O + (size_t)b * 2048 * 1024;

  const int t = threadIdx.x, wave = t >> 5, lane = t & 31;
  const int m = lane & 15, hi8 = (lane >> 4) << 3;

  // Persistent Q A-fragments: chunks c32 = wave + 8*f, f=0..3
  v16h qf[4];
#pragma unroll
  for (int f = 0; f < 4; ++f) {
    const h16* p = Qb + (size_t)(q0 + m) * 1024 + (wave + 8 * f) * 32 + hi8;
    v8h lo = *(const v8h*)p;
    v8h hi = *(const v8h*)(p + 16);
#pragma unroll
    for (int i = 0; i < 8; ++i) { qf[f][i] = lo[i]; qf[f][8 + i] = hi[i]; }
  }

  v8f o[8];
#pragma unroll
  for (int f = 0; f < 8; ++f)
#pragma unroll
    for (int i = 0; i < 8; ++i) o[f][i] = 0.0f;

  if (t < 16) { mrow[t] = -1e30f; lrow[t] = 0.0f; }

  // K staging: per-thread 4(kp) x 8(d) transpose block
  const int kg4 = (t >> 5) << 2;  // kp group: 0,4,...,28
  const int dg8 = (t & 31) << 3;  // d offset in quarter: 0..248
  // V staging: per-thread 16B chunk coordinates (4 async issues per quarter)
  // idx = t + 256*i, kp = idx>>5, dc = (idx&31)*8

  const int nblocks = (q0 + 16 + 31) >> 5;
  for (int blk = 0; blk < nblocks; ++blk) {
    const int kp0 = blk * 32;
    if (blk + 1 < nblocks)
      __builtin_prefetch(Kb + (size_t)(kp0 + 32 + kg4) * 1024 + dg8, 0, 3);

    v8f s0, s1;
#pragma unroll
    for (int i = 0; i < 8; ++i) { s0[i] = 0.0f; s1[i] = 0.0f; }

    // ---- QK^T over four 256-dim quarters; V quarters 0/1 DMA in background --
#pragma unroll
    for (int kq = 0; kq < 4; ++kq) {
      __syncthreads();  // Kbuf free; at kq==0 also: Vbufs free from prev block
      if (kq == 0) {
        // Kick off async DMA of V quarters 0 and 1 (4 instrs per wave each).
#pragma unroll
        for (int vq = 0; vq < 2; ++vq)
#pragma unroll
          for (int i = 0; i < 4; ++i) {
            int idx = t + 256 * i;
            int kp = idx >> 5, dc = (idx & 31) << 3;
            async_copy_b128(Vb + (size_t)(kp0 + kp) * 1024 + vq * 256 + dc,
                            &Vbuf[vq][kp * 256 + dc]);
          }
        Sc[2 * t] = 0.0f;
        Sc[2 * t + 1] = 0.0f;
      }
      {  // Stage K quarter transposed: Kbuf[d][kp], reg-blocked 4x8 transpose
        v8h rows[4];
#pragma unroll
        for (int r = 0; r < 4; ++r)
          rows[r] = *(const v8h*)(Kb + (size_t)(kp0 + kg4 + r) * 1024 + kq * 256 + dg8);
#pragma unroll
        for (int j = 0; j < 8; ++j) {
          v4h c;
#pragma unroll
          for (int r = 0; r < 4; ++r) c[r] = rows[r][j];
          *(v4h*)(Kbuf + (dg8 + j) * 32 + kg4) = c;
        }
      }
      __syncthreads();
      // Wave's chunk in this quarter sits at fixed offset wave*32 dims.
      v16h b0 = frag_b_load(Kbuf + (wave * 32) * 32, 32);
      v16h b1 = frag_b_load(Kbuf + (wave * 32) * 32 + 16, 32);
      s0 = WMMA_F32_F16(qf[kq], b0, s0);
      s1 = WMMA_F32_F16(qf[kq], b1, s1);
    }
    // Cross-wave reduction of partial scores
#pragma unroll
    for (int v = 0; v < 8; ++v) {
      atomicAdd(&Sc[(v + hi8) * 32 + (lane & 15)], s0[v]);
      atomicAdd(&Sc[(v + hi8) * 32 + 16 + (lane & 15)], s1[v]);
    }
    __syncthreads();

    // ---- Online softmax update (causal mask), 16 threads own 16 rows ----
    if (t < 16) {
      const int q = q0 + t;
      float mo = mrow[t], mx = mo;
#pragma unroll
      for (int j = 0; j < 32; ++j) {
        float s = (kp0 + j <= q) ? Sc[t * 32 + j] * 0.03125f : -1e30f;
        mx = fmaxf(mx, s);
      }
      float c = __expf(mo - mx);
      float ls = lrow[t] * c;
#pragma unroll
      for (int j = 0; j < 32; ++j) {
        float s = (kp0 + j <= q) ? Sc[t * 32 + j] * 0.03125f : -1e30f;
        float p = __expf(s - mx);
        ls += p;
        Ph[t * 32 + j] = (h16)p;
      }
      mrow[t] = mx; lrow[t] = ls; crow[t] = c;
    }
    __syncthreads();

    // Rescale running output by correction factor
    float cv[8];
#pragma unroll
    for (int v = 0; v < 8; ++v) cv[v] = crow[v + hi8];
#pragma unroll
    for (int f = 0; f < 8; ++f)
#pragma unroll
      for (int v = 0; v < 8; ++v) o[f][v] *= cv[v];

    v16h pf = frag_a_load(Ph, 32);  // P as 16x32 A fragment

    // ---- PV over four V quarters (double-buffered async pipeline) ----
#pragma unroll
    for (int vq = 0; vq < 4; ++vq) {
      if (vq < 3) async_wait4(); else async_wait0();  // quarter vq landed
      __syncthreads();
      const h16* vb = Vbuf[vq & 1];
#pragma unroll
      for (int j = 0; j < 2; ++j) {
        v16h vfrag = frag_b_load(vb + wave * 32 + j * 16, 256);
        o[vq * 2 + j] = WMMA_F32_F16(pf, vfrag, o[vq * 2 + j]);
      }
      if (vq < 2) {  // refill this buffer with quarter vq+2
        __syncthreads();  // all waves done reading Vbuf[vq&1]
#pragma unroll
        for (int i = 0; i < 4; ++i) {
          int idx = t + 256 * i;
          int kp = idx >> 5, dc = (idx & 31) << 3;
          async_copy_b128(Vb + (size_t)(kp0 + kp) * 1024 + (vq + 2) * 256 + dc,
                          &Vbuf[vq & 1][kp * 256 + dc]);
        }
      }
    }
  }
  __syncthreads();

  // Final normalization and fp32 store
  float iv[8];
#pragma unroll
  for (int v = 0; v < 8; ++v) iv[v] = 1.0f / lrow[v + hi8];
#pragma unroll
  for (int f = 0; f < 8; ++f) {
    int c32 = wave + 8 * (f >> 1);
    int col = c32 * 32 + (f & 1) * 16 + (lane & 15);
#pragma unroll
    for (int v = 0; v < 8; ++v)
      Ob[(size_t)(q0 + v + hi8) * 1024 + col] = o[f][v] * iv[v];
  }
}

extern "C" void kernel_launch(void* const* d_in, const int* in_sizes, int n_in,
                              void* d_out, int out_size, void* d_ws, size_t ws_size,
                              hipStream_t stream) {
  const float* x  = (const float*)d_in[0];
  const float* Wq = (const float*)d_in[1];
  const float* Wk = (const float*)d_in[2];
  const float* Wv = (const float*)d_in[3];
  float* out = (float*)d_out;

  // Workspace: f16 Q, K, V — 8192x1024 each (16 MB each, 48 MB total)
  h16* Qh = (h16*)d_ws;
  h16* Kh = Qh + (size_t)8192 * 1024;
  h16* Vh = Kh + (size_t)8192 * 1024;

  qkv_proj<<<dim3(64, 8, 3), 256, 0, stream>>>(x, Wq, Wk, Wv, Qh, Kh, Vh);
  flash_attn<<<dim3(128, 4), 256, 0, stream>>>(Qh, Kh, Vh, out);
}
